// VanillaRNN_18425409700270
// MI455X (gfx1250) — compile-verified, matrix operand order checked by reading, not run on hardware
//
#include <hip/hip_runtime.h>
#include <cmath>

typedef __attribute__((ext_vector_type(16))) _Float16 v16h;
typedef __attribute__((ext_vector_type(8)))  float    v8f;
typedef __attribute__((ext_vector_type(4)))  float    f32x4;

#define T_LEN 200
#define F_IN  4
#define H_DIM 10

__device__ __forceinline__ float fast_tanh(float v) {
#if __has_builtin(__builtin_amdgcn_tanhf)
    return __builtin_amdgcn_tanhf(v);   // gfx1250 v_tanh_f32 (TRANS op)
#else
    return tanhf(v);
#endif
}

// One wave32 owns 16 batch columns. All matrices are computed TRANSPOSED
// (D = W * h^T), so the WMMA D-layout (rows striped across lanes) coincides
// with the B-operand layout of the next step: the recurrence feeds back with
// per-lane f32->f16 packing only, no cross-lane shuffles.
//
//   A0    = [W_hh0 | W_ih0 | 0]  (16x32 f16: K=0..9 recurrent, K=10..13 input)
//   B0    = [h0^T ; x_t^T ; 0]   (32x16 f16)
//   d0    = A0*B0 + bias0        -> h0_new = tanh(d0)   (1 WMMA, inproj fused)
//   dt    = A_hh1*B1 + bias1                            (independent WMMA)
//   d1    = A_ih1*B0_new + dt    -> h1_new = tanh(d1)
//   acc  += sum_m W_fc[t*10+m] * h1_new[m]   (FC head fused, ys never stored)
__global__ __launch_bounds__(256) void rnn2_fused_wmma(
    const float* __restrict__ x,
    const float* __restrict__ Wih0, const float* __restrict__ Whh0,
    const float* __restrict__ bih0, const float* __restrict__ bhh0,
    const float* __restrict__ Wih1, const float* __restrict__ Whh1,
    const float* __restrict__ bih1, const float* __restrict__ bhh1,
    const float* __restrict__ Wfc,  const float* __restrict__ bfc,
    float* __restrict__ out, int batch)
{
    const int tid  = blockIdx.x * blockDim.x + threadIdx.x;
    const int wave = tid >> 5;
    const int lane = tid & 31;
    const int hi   = lane >> 4;        // 0: lanes 0-15, 1: lanes 16-31
    const int row  = lane & 15;        // A-matrix row / batch column within tile
    const int kb   = hi * 8;           // K-base this lane half holds (A and B)
    const int col  = wave * 16 + row;  // global batch index of this lane's column

    // ---- stage FC weights into LDS, zero-padded to 16 per timestep ----
    // sWfc[t*16 + m] = (m < 10) ? Wfc[t*10 + m] : 0
    // -> inner loop does two aligned ds_load_b128 per step, no predication.
    __shared__ float sWfc[T_LEN * 16];
    for (int i = threadIdx.x; i < T_LEN * 16; i += blockDim.x) {
        const int t = i >> 4, m = i & 15;
        sWfc[i] = (m < H_DIM) ? Wfc[t * H_DIM + m] : 0.f;
    }

    // ---- build constant A operands (16x32 f16, zero padded) ----
    v16h A0, Aih1, Ahh1;
    for (int j = 0; j < 16; ++j) {
        A0[j] = (_Float16)0.f; Aih1[j] = (_Float16)0.f; Ahh1[j] = (_Float16)0.f;
    }
    if (row < H_DIM) {
        for (int j = 0; j < 8; ++j) {
            const int k = kb + j;
            if (k < H_DIM) {
                A0[j]   = (_Float16)Whh0[row * H_DIM + k];
                Aih1[j] = (_Float16)Wih1[row * H_DIM + k];
                Ahh1[j] = (_Float16)Whh1[row * H_DIM + k];
            } else if (k < H_DIM + F_IN) {
                A0[j]   = (_Float16)Wih0[row * F_IN + (k - H_DIM)];  // fused inproj
            }
        }
    }

    // ---- bias accumulators in C/D layout: VGPR r holds row m = r + 8*hi ----
    v8f C0, C1;
    for (int r = 0; r < 8; ++r) {
        const int m = r + 8 * hi;
        float b0v = 0.f, b1v = 0.f;
        if (m < H_DIM) { b0v = bih0[m] + bhh0[m]; b1v = bih1[m] + bhh1[m]; }
        C0[r] = b0v; C1[r] = b1v;
    }

    // ---- state in B-operand layout (h = 0 initially) ----
    v16h B0, B1;
    for (int j = 0; j < 16; ++j) { B0[j] = (_Float16)0.f; B1[j] = (_Float16)0.f; }

    float acc = 0.f;
    const float* __restrict__ xp = x + (size_t)col * (T_LEN * F_IN);

    __syncthreads();   // sWfc ready

    for (int t = 0; t < T_LEN; ++t) {
        // x_t for this column; 16B aligned (col*800 + t*4 floats)
        const f32x4 xv = *(const f32x4*)(xp + t * F_IN);
        if (t + 8 < T_LEN) __builtin_prefetch(xp + (t + 8) * F_IN, 0, 1);

        // Insert x_t into B0's K=10..13 slots (live in hi-half elements 2..5).
        if (hi) {
            B0[2] = (_Float16)xv.x; B0[3] = (_Float16)xv.y;
            B0[4] = (_Float16)xv.z; B0[5] = (_Float16)xv.w;
        }

        // layer 0 (recurrent + input projection in one WMMA), and the
        // independent W_hh1*h1 WMMA for ILP across the tanh latency.
        v8f d0 = __builtin_amdgcn_wmma_f32_16x16x32_f16(
                     false, A0,   false, B0, (short)0, C0, false, false);
        v8f dt = __builtin_amdgcn_wmma_f32_16x16x32_f16(
                     false, Ahh1, false, B1, (short)0, C1, false, false);

        // h0 = tanh(d0); pack back into B layout (pure per-lane).
        // Padded rows are exactly 0 (zero A rows + zero bias) -> tanh(0)=0.
        for (int j = 0; j < 8; ++j) B0[j] = (_Float16)fast_tanh(d0[j]);

        v8f d1 = __builtin_amdgcn_wmma_f32_16x16x32_f16(
                     false, Aih1, false, B0, (short)0, dt, false, false);

        // h1 = tanh(d1); pack, and fold FC-head contribution for step t.
        // Padded rows (m >= 10) multiply the zero-padded LDS weights.
        const float* __restrict__ wrow = &sWfc[t * 16 + kb];
        for (int j = 0; j < 8; ++j) {
            const float h = fast_tanh(d1[j]);
            B1[j] = (_Float16)h;
            acc = fmaf(wrow[j], h, acc);
        }
    }

    // Combine the two lane halves (m=0..7 partial + m=8..9 partial).
    acc += __shfl_xor(acc, 16, 32);
    if (hi == 0) {
        const float z = acc + bfc[0];
        out[col] = 1.f / (1.f + __expf(-z));
    }
}

extern "C" void kernel_launch(void* const* d_in, const int* in_sizes, int n_in,
                              void* d_out, int out_size, void* d_ws, size_t ws_size,
                              hipStream_t stream) {
    const float* x    = (const float*)d_in[0];
    const float* Wih0 = (const float*)d_in[1];
    const float* Whh0 = (const float*)d_in[2];
    const float* bih0 = (const float*)d_in[3];
    const float* bhh0 = (const float*)d_in[4];
    const float* Wih1 = (const float*)d_in[5];
    const float* Whh1 = (const float*)d_in[6];
    const float* bih1 = (const float*)d_in[7];
    const float* bhh1 = (const float*)d_in[8];
    const float* Wfc  = (const float*)d_in[9];
    const float* bfc  = (const float*)d_in[10];
    float* out = (float*)d_out;

    const int batch = in_sizes[0] / (T_LEN * F_IN);   // 16384
    const int waves = (batch + 15) / 16;              // 1024 tiles of 16 columns
    const int threads = 256;                          // 8 waves per block
    const int blocks = (waves * 32 + threads - 1) / threads;  // 128

    rnn2_fused_wmma<<<blocks, threads, 0, stream>>>(
        x, Wih0, Whh0, bih0, bhh0, Wih1, Whh1, bih1, bhh1, Wfc, bfc, out, batch);
}
